// EfficientAttention_75548474737234
// MI455X (gfx1250) — compile-verified, matrix operand order checked by reading, not run on hardware
//
#include <hip/hip_runtime.h>
#include <hip/hip_bf16.h>
#include <stdint.h>

typedef __attribute__((ext_vector_type(16))) __bf16 bf16x16;
typedef __attribute__((ext_vector_type(8)))  float  v8f;
typedef __attribute__((ext_vector_type(4)))  int    v4i;

#define C_DIM   256
#define L_DIM   196
#define LP      208
#define N_PATCH 1024
#define HW      224

// ---------- bf16 helpers ----------
static __device__ __forceinline__ unsigned short f2bf_us(float f) {
  unsigned u = __float_as_uint(f);
  unsigned r = u + 0x7fffu + ((u >> 16) & 1u);   // round-to-nearest-even
  return (unsigned short)(r >> 16);
}
static __device__ __forceinline__ __bf16 us_bf(unsigned short s) {
  return __builtin_bit_cast(__bf16, s);
}
static __device__ __forceinline__ __bf16 lo_bf(unsigned u){ return us_bf((unsigned short)(u & 0xffffu)); }
static __device__ __forceinline__ __bf16 hi_bf(unsigned u){ return us_bf((unsigned short)(u >> 16)); }

static __device__ __forceinline__ v8f zero8() {
  v8f r;
  #pragma unroll
  for (int i = 0; i < 8; ++i) r[i] = 0.0f;
  return r;
}

static __device__ __forceinline__ v8f wmma_bf16(bf16x16 a, bf16x16 b, v8f c) {
  // D = A(16x32 bf16) * B(32x16 bf16) + C(16x16 f32)
  return __builtin_amdgcn_wmma_f32_16x16x32_bf16(false, a, false, b, (short)0, c, false, false);
}

// ---------- async global->LDS staging (CDNA5 path, guarded fallback) ----------
#if defined(__has_builtin)
#if __has_builtin(__builtin_amdgcn_global_load_async_to_lds_b128)
#define HAVE_ASYNC_LDS 1
#endif
#endif

#ifdef HAVE_ASYNC_LDS
typedef __attribute__((address_space(1))) v4i g_v4i;   // global (device) int4
typedef __attribute__((address_space(3))) v4i l_v4i;   // LDS int4
#endif

static __device__ __forceinline__ void async_cp16(const unsigned short* gsrc, unsigned short* lds) {
#ifdef HAVE_ASYNC_LDS
  __builtin_amdgcn_global_load_async_to_lds_b128((g_v4i*)gsrc, (l_v4i*)lds, 0, 0);
#else
  *(uint4*)lds = *(const uint4*)gsrc;
#endif
}

static __device__ __forceinline__ void async_wait() {
#ifdef HAVE_ASYNC_LDS
#if __has_builtin(__builtin_amdgcn_s_wait_asynccnt)
  __builtin_amdgcn_s_wait_asynccnt(0);
#else
  asm volatile("s_wait_asynccnt 0x0" ::: "memory");
#endif
#endif
}

// ---------- WMMA operand loaders ----------
// 16x32 operand tile from a row-major bf16 matrix (A: rows=M over lanes; B: rows=N over lanes).
// ISA 16-bit operand layout: lane half selects K {0..7,16..23} vs {8..15,24..31}; vgpr v holds
// K pair (half*8 + (v&3)*2 + (v>>2)*16) -> two contiguous 16B chunks per lane.
static __device__ __forceinline__ bf16x16 ld_rm(const unsigned short* p, int stride, int row0, int k0) {
  int lane = threadIdx.x & 31, half = lane >> 4, ln = lane & 15;
  const unsigned* rp = (const unsigned*)(p + (size_t)(row0 + ln) * stride + k0 + half * 8);
  bf16x16 r;
  #pragma unroll
  for (int v = 0; v < 8; ++v) {
    int koff = ((v & 3) << 1) + ((v >> 2) << 4);   // 0,2,4,6,16,18,20,22
    unsigned u = rp[koff >> 1];
    r[2*v]   = lo_bf(u);
    r[2*v+1] = hi_bf(u);
  }
  return r;
}

// Operand whose K-dimension runs down rows: element k at p[k*rowstride + col].
static __device__ __forceinline__ bf16x16 ld_cm(const unsigned short* p, int rowstride, int col0) {
  int lane = threadIdx.x & 31, half = lane >> 4, ln = lane & 15;
  const unsigned short* cp = p + col0 + ln;
  bf16x16 r;
  #pragma unroll
  for (int v = 0; v < 8; ++v) {
    int koff = half * 8 + ((v & 3) << 1) + ((v >> 2) << 4);
    r[2*v]   = us_bf(cp[(size_t)koff * rowstride]);
    r[2*v+1] = us_bf(cp[(size_t)(koff + 1) * rowstride]);
  }
  return r;
}

// ---------- kernel 1: (bilinear resize) + patchify + fp32->bf16, transposed layout [n][Lp][C] ----------
__global__ __launch_bounds__(256) void k_patchify(const float* __restrict__ src, int Hs,
                                                  unsigned short* __restrict__ dst) {
  size_t idx = (size_t)blockIdx.x * 256 + threadIdx.x;
  const size_t total = (size_t)N_PATCH * LP * C_DIM;
  if (idx >= total) return;
  int c = (int)(idx % C_DIM);
  int l = (int)((idx / C_DIM) % LP);
  int n = (int)(idx / ((size_t)C_DIM * LP));
  unsigned short o = 0;
  if (l < L_DIM) {
    int b = n >> 8, pidx = n & 255, ph = pidx >> 4, pw = pidx & 15;
    int h = ph * 14 + l / 14, w = pw * 14 + l % 14;
    float v;
    if (Hs == HW) {
      v = src[(((size_t)b * C_DIM + c) * HW + h) * HW + w];
    } else {
      float sc = (float)Hs / (float)HW;
      float fy = ((float)h + 0.5f) * sc - 0.5f;
      float fx = ((float)w + 0.5f) * sc - 0.5f;
      int y0 = (int)floorf(fy), x0 = (int)floorf(fx);
      float dy = fy - (float)y0, dx = fx - (float)x0;
      int y1 = y0 + 1, x1 = x0 + 1;
      y0 = y0 < 0 ? 0 : (y0 > Hs - 1 ? Hs - 1 : y0);
      y1 = y1 < 0 ? 0 : (y1 > Hs - 1 ? Hs - 1 : y1);
      x0 = x0 < 0 ? 0 : (x0 > Hs - 1 ? Hs - 1 : x0);
      x1 = x1 < 0 ? 0 : (x1 > Hs - 1 ? Hs - 1 : x1);
      const float* sp = src + ((size_t)b * C_DIM + c) * Hs * Hs;
      float v00 = sp[(size_t)y0 * Hs + x0], v01 = sp[(size_t)y0 * Hs + x1];
      float v10 = sp[(size_t)y1 * Hs + x0], v11 = sp[(size_t)y1 * Hs + x1];
      v = (1.f - dy) * ((1.f - dx) * v00 + dx * v01) + dy * ((1.f - dx) * v10 + dx * v11);
    }
    o = f2bf_us(v);
  }
  dst[idx] = o;
}

// ---------- kernel 2: projection GEMM  Yt[n][l][c] = sum_k Xt[n][l][k] * W[c][k] + bias[c] ----------
// grid = N_PATCH * 13 (one 16-row L tile per WG), 256 threads = 8 waves, 2 c-tiles per wave.
__global__ __launch_bounds__(256) void k_proj(const unsigned short* __restrict__ Xt,
                                              const float* __restrict__ W,
                                              const float* __restrict__ bias,
                                              unsigned short* __restrict__ Yt) {
  __shared__ __attribute__((aligned(16))) unsigned short Xs[16 * C_DIM];   // 8 KB
  __shared__ __attribute__((aligned(16))) unsigned short Wls[C_DIM * 32];  // 16 KB
  int n = blockIdx.x / 13, mt = blockIdx.x % 13, l0 = mt * 16;
  int t = threadIdx.x, wave = t >> 5, lane = t & 31, half = lane >> 4, ln = lane & 15;

  // stage X tile (512 x 16B chunks, async -> LDS)
  {
    const unsigned short* xsrc = Xt + ((size_t)n * LP + l0) * C_DIM;
    #pragma unroll
    for (int i = 0; i < 2; ++i) {
      int u = t + i * 256;
      async_cp16(xsrc + u * 8, Xs + u * 8);
    }
    async_wait();
  }

  v8f acc0 = zero8(), acc1 = zero8();
  int ct0 = wave * 2, ct1 = wave * 2 + 1;

  for (int kk = 0; kk < 8; ++kk) {
    __syncthreads();
    { // stage W[:, kk*32 .. kk*32+31] as bf16; thread t owns row t (needs f32->bf16 convert)
      const float* wr = W + (size_t)t * C_DIM + kk * 32;
      #pragma unroll
      for (int j = 0; j < 32; ++j) Wls[t * 32 + j] = f2bf_us(wr[j]);
    }
    __syncthreads();
    bf16x16 a  = ld_rm(Xs, C_DIM, 0, kk * 32);
    bf16x16 b0 = ld_rm(Wls, 32, ct0 * 16, 0);
    bf16x16 b1 = ld_rm(Wls, 32, ct1 * 16, 0);
    acc0 = wmma_bf16(a, b0, acc0);
    acc1 = wmma_bf16(a, b1, acc1);
  }

  // epilogue: D element (vgpr j, lane) -> l = l0 + j + 8*half, c = ct*16 + ln
  float bv0 = bias[ct0 * 16 + ln];
  float bv1 = bias[ct1 * 16 + ln];
  #pragma unroll
  for (int j = 0; j < 8; ++j) {
    int l = l0 + j + 8 * half;
    Yt[((size_t)n * LP + l) * C_DIM + ct0 * 16 + ln] = f2bf_us(acc0[j] + bv0);
    Yt[((size_t)n * LP + l) * C_DIM + ct1 * 16 + ln] = f2bf_us(acc1[j] + bv1);
  }
}

// ---------- kernel 3: attention  out += V * softmax(Qt Kt^T)^T, fused unpatchify ----------
// grid = N_PATCH * 4 (4 L-blocks per WG, one per wave), 128 threads = 4 waves.
__global__ __launch_bounds__(128) void k_attn(const unsigned short* __restrict__ Qt,
                                              const unsigned short* __restrict__ Kt,
                                              const unsigned short* __restrict__ Vt,
                                              float* __restrict__ out, int beta) {
  __shared__ __attribute__((aligned(16))) unsigned short Ks[16 * C_DIM];   // 8 KB
  __shared__ __attribute__((aligned(16))) unsigned short Vs[32 * C_DIM];   // 16 KB
  __shared__ __attribute__((aligned(16))) unsigned short Ps[4][16 * 224];  // 28 KB
  int n = blockIdx.x >> 2, g = blockIdx.x & 3;
  int t = threadIdx.x, wave = t >> 5, lane = t & 31, half = lane >> 4, ln = lane & 15;
  int lb = g * 4 + wave;
  bool active = lb < 13;
  int l0 = lb * 16;

  // preload Q row-block operands (reused across all 13 m-tiles)
  bf16x16 q[8];
  if (active) {
    #pragma unroll
    for (int kk = 0; kk < 8; ++kk)
      q[kk] = ld_rm(Qt + (size_t)n * LP * C_DIM, C_DIM, l0, kk * 32);
  }

  v8f sacc[13];
  #pragma unroll
  for (int mt = 0; mt < 13; ++mt) sacc[mt] = zero8();

  // zero P padding columns m=208..223 for this wave
  if (active) {
    #pragma unroll
    for (int j = 0; j < 8; ++j) Ps[wave][(j + 8 * half) * 224 + 208 + ln] = 0;
  }

  // ---- phase 1: S = Qt * Kt^T, S tiles live in registers ----
  #pragma unroll
  for (int mt = 0; mt < 13; ++mt) {
    __syncthreads();
    { // stage Kt rows [mt*16, mt*16+16): 512 x 16B chunks, async -> LDS
      const unsigned short* ksrc = Kt + ((size_t)n * LP + mt * 16) * C_DIM;
      #pragma unroll
      for (int i = 0; i < 4; ++i) {
        int u = t + i * 128;
        async_cp16(ksrc + u * 8, Ks + u * 8);
      }
      async_wait();
    }
    __syncthreads();
    if (active) {
      #pragma unroll
      for (int kk = 0; kk < 8; ++kk) {
        bf16x16 b = ld_rm(Ks, C_DIM, 0, kk * 32);
        sacc[mt] = wmma_bf16(q[kk], b, sacc[mt]);
      }
    }
  }

  // ---- phase 2: masked row softmax in registers (row = (j, half); cols across lanes 0..15) ----
  if (active) {
    #pragma unroll
    for (int j = 0; j < 8; ++j) {
      float mx = -3.0e38f;
      #pragma unroll
      for (int mt = 0; mt < 13; ++mt) {
        int m = mt * 16 + ln;
        if (m < L_DIM) mx = fmaxf(mx, sacc[mt][j]);
      }
      #pragma unroll
      for (int s = 1; s < 16; s <<= 1) mx = fmaxf(mx, __shfl_xor(mx, s));
      float sum = 0.f;
      #pragma unroll
      for (int mt = 0; mt < 13; ++mt) {     // exp in place: sacc <- exp(s - max), masked -> 0
        int m = mt * 16 + ln;
        float e = (m < L_DIM) ? __expf(sacc[mt][j] - mx) : 0.0f;
        sacc[mt][j] = e;
        sum += e;
      }
      #pragma unroll
      for (int s = 1; s < 16; s <<= 1) sum += __shfl_xor(sum, s);
      float inv = 1.0f / sum;
      int lrow = (j + 8 * half) * 224;
      #pragma unroll
      for (int mt = 0; mt < 13; ++mt) {
        int m = mt * 16 + ln;
        Ps[wave][lrow + m] = f2bf_us(sacc[mt][j] * inv);
      }
    }
  }

  // ---- phase 3: out(c, l-block) = V * P^T ; K-dim = m padded to 224 ----
  v8f oacc[16];
  #pragma unroll
  for (int ct = 0; ct < 16; ++ct) oacc[ct] = zero8();

  #pragma unroll
  for (int ks = 0; ks < 7; ++ks) {
    int m0 = ks * 32;
    __syncthreads();
    { // stage Vt rows [m0, m0+min(32, LP-m0)) async; zero-fill the first 8 tail rows of the last slab
      int rows = (m0 + 32 <= LP) ? 32 : (LP - m0);
      int chunks = rows * 32;                    // 32 x 16B chunks per 256-elem row
      const unsigned short* vsrc = Vt + ((size_t)n * LP + m0) * C_DIM;
      #pragma unroll
      for (int i = 0; i < 8; ++i) {
        int u = t + i * 128;
        if (u < chunks) async_cp16(vsrc + u * 8, Vs + u * 8);
      }
      if (rows < 32) {                           // zero rows [rows, rows+8): 1024 dwords
        unsigned* z = (unsigned*)(Vs + rows * C_DIM);
        #pragma unroll
        for (int i = 0; i < 8; ++i) z[t + i * 128] = 0u;
      }
      async_wait();
    }
    __syncthreads();
    if (active) {
      bf16x16 b = ld_rm(&Ps[wave][0], 224, 0, m0);   // N=l over lanes, K=m
      #pragma unroll
      for (int ct = 0; ct < 16; ++ct) {
        bf16x16 a = ld_cm(Vs, C_DIM, ct * 16);       // M=c over lanes, K=m down rows
        oacc[ct] = wmma_bf16(a, b, oacc[ct]);
      }
    }
  }
  __syncthreads();

  // ---- epilogue: unpatchify + accumulate over skips ----
  if (active) {
    int l = l0 + ln;
    if (l < L_DIM) {
      int b_ = n >> 8, pidx = n & 255, ph = pidx >> 4, pw = pidx & 15;
      int h = ph * 14 + l / 14, w = pw * 14 + l % 14;
      size_t base = ((size_t)b_ * C_DIM) * HW * HW + (size_t)h * HW + w;
      #pragma unroll
      for (int ct = 0; ct < 16; ++ct) {
        #pragma unroll
        for (int j = 0; j < 8; ++j) {
          int c = ct * 16 + j + 8 * half;
          size_t idx = base + (size_t)c * HW * HW;
          float v = oacc[ct][j];
          if (beta) v += out[idx];
          out[idx] = v;
        }
      }
    }
  }
}

// ---------- host launcher ----------
extern "C" void kernel_launch(void* const* d_in, const int* in_sizes, int n_in,
                              void* d_out, int out_size, void* d_ws, size_t ws_size,
                              hipStream_t stream) {
  (void)in_sizes; (void)n_in; (void)out_size; (void)ws_size;
  const float* x     = (const float*)d_in[0];
  const float* skip0 = (const float*)d_in[1];
  const float* skip1 = (const float*)d_in[2];
  const float* Wq = (const float*)d_in[3]; const float* bq = (const float*)d_in[4];
  const float* Wk = (const float*)d_in[5]; const float* bk = (const float*)d_in[6];
  const float* Wv = (const float*)d_in[7]; const float* bv = (const float*)d_in[8];
  float* out = (float*)d_out;

  unsigned short* ws = (unsigned short*)d_ws;
  const size_t PB = (size_t)N_PATCH * LP * C_DIM;     // elements per buffer
  unsigned short* bufX = ws;
  unsigned short* bufQ = ws + PB;
  unsigned short* bufK = ws + 2 * PB;
  unsigned short* bufV = ws + 3 * PB;

  const int gPatch = (int)((PB + 255) / 256);          // 212992
  const int gProj  = N_PATCH * 13;                     // 13312
  const int gAttn  = N_PATCH * 4;                      // 4096

  // Q = Wq * patchify(x) + bq   (shared across skips)
  k_patchify<<<gPatch, 256, 0, stream>>>(x, HW, bufX);
  k_proj<<<gProj, 256, 0, stream>>>(bufX, Wq, bq, bufQ);

  // skip0 (112 -> 224 bilinear)
  k_patchify<<<gPatch, 256, 0, stream>>>(skip0, 112, bufX);
  k_proj<<<gProj, 256, 0, stream>>>(bufX, Wk, bk, bufK);
  k_proj<<<gProj, 256, 0, stream>>>(bufX, Wv, bv, bufV);
  k_attn<<<gAttn, 128, 0, stream>>>(bufQ, bufK, bufV, out, 0);

  // skip1 (56 -> 224 bilinear)
  k_patchify<<<gPatch, 256, 0, stream>>>(skip1, 56, bufX);
  k_proj<<<gProj, 256, 0, stream>>>(bufX, Wk, bk, bufK);
  k_proj<<<gProj, 256, 0, stream>>>(bufX, Wv, bv, bufV);
  k_attn<<<gAttn, 128, 0, stream>>>(bufQ, bufK, bufV, out, 1);
}
